// PromptQueryFusionBlock_5875515261610
// MI455X (gfx1250) — compile-verified, matrix operand order checked by reading, not run on hardware
//
#include <hip/hip_runtime.h>
#include <hip/hip_bf16.h>
#include <math.h>

// Problem dims
#define BB 8
#define CC 256
#define HH 64
#define WW 64
#define NN (HH*WW)     // 4096 tokens
#define PP 32          // prompt tokens
#define KCLS 16        // classes
#define PDIM 256
#define NHEAD 8
#define HD (CC/NHEAD)  // 32
#define CIN (CC+4)     // 260 conv input channels
#define CINP 288       // padded to 9*32
#define CP2 (CINP/2)   // 144 bf16 pairs

typedef __attribute__((ext_vector_type(16))) __bf16 v16bf;
typedef __attribute__((ext_vector_type(8)))  float  v8f;
typedef __attribute__((ext_vector_type(8)))  unsigned v8u;

// Pack two f32 into one dword of two bf16 (round-half-up): 1 add each + 1 v_perm_b32
__device__ inline unsigned pack_bf16_pair(float f0, float f1) {
  unsigned u0 = __builtin_bit_cast(unsigned, f0) + 0x8000u;
  unsigned u1 = __builtin_bit_cast(unsigned, f1) + 0x8000u;
  return __builtin_amdgcn_perm(u1, u0, 0x07060302u);
}

__device__ inline float wave_sum(float s) {
#pragma unroll
  for (int o = 16; o > 0; o >>= 1) s += __shfl_xor(s, o, 32);
  return s;
}

__device__ inline float gelu_f(float x) {
  float x3 = x * x * x;
  return 0.5f * x * (1.f + tanhf(0.7978845608028654f * (x + 0.044715f * x3)));
}

// ---- WMMA bf16 fragment loaders (wave32 layouts per ISA 7.12.2), branch-free ----
// A-style tile (row = lane&15, K contiguous). Requires base+lda 8B-aligned, K%32==0.
__device__ inline v16bf load_a_frag(const float* __restrict__ A, int lda, int k0, int lane) {
  int m = lane & 15, kg = lane >> 4;
  const float* row = A + (size_t)m * lda + k0 + 8 * kg;
  v8u u;
#pragma unroll
  for (int v = 0; v < 8; ++v) {
    int off = 2 * v + ((v >= 4) ? 8 : 0);   // 0,2,4,6,16,18,20,22
    float2 f = *(const float2*)(row + off);
    u[v] = pack_bf16_pair(f.x, f.y);
  }
  return __builtin_bit_cast(v16bf, u);
}
// B tile, non-transposed [K,N] row-major: col = lane&15, strided dword loads.
__device__ inline v16bf load_bn_frag(const float* __restrict__ Bm, int ldb, int k0, int lane) {
  int n = lane & 15, kg = lane >> 4;
  const float* base = Bm + n + (size_t)(k0 + 8 * kg) * ldb;
  v8u u;
#pragma unroll
  for (int v = 0; v < 8; ++v) {
    int kb = 2 * v + ((v >= 4) ? 8 : 0);
    float f0 = base[(size_t)kb * ldb];
    float f1 = base[(size_t)(kb + 1) * ldb];
    u[v] = pack_bf16_pair(f0, f1);
  }
  return __builtin_bit_cast(v16bf, u);
}

// ---------------- generic batched GEMM: C = epi(A @ B * scale + bias) ----------------
// K must be a multiple of 32; M%16==0; N%32==0.
struct Gemm {
  const float* A; const float* Bm; float* C;
  const float* bias;
  long aOuter, aInner, bOuter, bInner, cOuter, cInner;
  int lda, ldb, ldc;
  int M, N, K;
  int outer, inner;
  float scale;
};

template <int BT, int EPI>
__global__ void gemm_wmma_kernel(Gemm g) {
  int wave = (blockIdx.x * blockDim.x + threadIdx.x) >> 5;
  int lane = threadIdx.x & 31;
  int mt  = g.M >> 4;
  int nt2 = g.N >> 5;
  long tiles = (long)g.outer * g.inner * mt * nt2;
  if (wave >= tiles) return;
  long t = wave;
  int n2 = (int)(t % nt2); t /= nt2;
  int mi = (int)(t % mt);  t /= mt;
  int in = (int)(t % g.inner); t /= g.inner;
  int ou = (int)t;

  const float* A  = g.A + (long)ou * g.aOuter + (long)in * g.aInner + (size_t)(mi * 16) * g.lda;
  const float* Bb = g.Bm + (long)ou * g.bOuter + (long)in * g.bInner;
  const float* B0 = BT ? Bb + (size_t)(n2 * 32) * g.ldb      : Bb + n2 * 32;
  const float* B1 = BT ? Bb + (size_t)(n2 * 32 + 16) * g.ldb : Bb + n2 * 32 + 16;
  float* C = g.C + (long)ou * g.cOuter + (long)in * g.cInner + (size_t)(mi * 16) * g.ldc + n2 * 32;

  v8f acc0 = {}; v8f acc1 = {};
  for (int k0 = 0; k0 < g.K; k0 += 32) {
    v16bf a = load_a_frag(A, g.lda, k0, lane);
    v16bf b0, b1;
    if (BT) { b0 = load_a_frag(B0, g.ldb, k0, lane);   // B^T tile has A-style layout
              b1 = load_a_frag(B1, g.ldb, k0, lane); }
    else    { b0 = load_bn_frag(B0, g.ldb, k0, lane);
              b1 = load_bn_frag(B1, g.ldb, k0, lane); }
    acc0 = __builtin_amdgcn_wmma_f32_16x16x32_bf16(false, a, false, b0, (short)0, acc0, false, false);
    acc1 = __builtin_amdgcn_wmma_f32_16x16x32_bf16(false, a, false, b1, (short)0, acc1, false, false);
  }

  int n = lane & 15, mh = lane >> 4;
  int colBase = n2 * 32;
  float bias0 = g.bias ? g.bias[colBase + n] : 0.f;
  float bias1 = g.bias ? g.bias[colBase + 16 + n] : 0.f;
#pragma unroll
  for (int r = 0; r < 8; ++r) {
    int m = r + 8 * mh;
    float v0 = acc0[r] * g.scale + bias0;
    float v1 = acc1[r] * g.scale + bias1;
    float* c0 = C + (size_t)m * g.ldc + n;
    float* c1 = c0 + 16;
    if (EPI == 0)      { *c0 = v0;          *c1 = v1; }
    else if (EPI == 1) { *c0 = gelu_f(v0);  *c1 = gelu_f(v1); }
    else               { *c0 += v0;         *c1 += v1; }
  }
}

// ---------------- LayerNorm over C for query (reads x[b,c,n], writes [b,n,c]) ----------------
__global__ void ln_query_kernel(const float* __restrict__ x, const float* __restrict__ g,
                                const float* __restrict__ b, float* __restrict__ out) {
  int wave = (blockIdx.x * blockDim.x + threadIdx.x) >> 5;
  int lane = threadIdx.x & 31;
  if (wave >= BB * NN) return;
  int bi = wave / NN, n = wave % NN;
  const float* xb = x + (size_t)bi * CC * NN + n;
  float vals[CC / 32];
  float s = 0.f;
#pragma unroll
  for (int i = 0; i < CC / 32; ++i) { int c = lane + 32 * i; vals[i] = xb[(size_t)c * NN]; s += vals[i]; }
  float mu = wave_sum(s) * (1.f / CC);
  float vs = 0.f;
#pragma unroll
  for (int i = 0; i < CC / 32; ++i) { float d = vals[i] - mu; vs += d * d; }
  float inv = rsqrtf(wave_sum(vs) * (1.f / CC) + 1e-5f);
  float* ob = out + (size_t)wave * CC;
#pragma unroll
  for (int i = 0; i < CC / 32; ++i) { int c = lane + 32 * i; ob[c] = (vals[i] - mu) * inv * g[c] + b[c]; }
}

// ---------------- LayerNorm for contiguous rows [rows, C] ----------------
__global__ void ln_rows_kernel(const float* __restrict__ in, const float* __restrict__ g,
                               const float* __restrict__ b, float* __restrict__ out, int rows) {
  int wave = (blockIdx.x * blockDim.x + threadIdx.x) >> 5;
  int lane = threadIdx.x & 31;
  if (wave >= rows) return;
  const float* r = in + (size_t)wave * CC;
  float vals[CC / 32];
  float s = 0.f;
#pragma unroll
  for (int i = 0; i < CC / 32; ++i) { int c = lane + 32 * i; vals[i] = r[c]; s += vals[i]; }
  float mu = wave_sum(s) * (1.f / CC);
  float vs = 0.f;
#pragma unroll
  for (int i = 0; i < CC / 32; ++i) { float d = vals[i] - mu; vs += d * d; }
  float inv = rsqrtf(wave_sum(vs) * (1.f / CC) + 1e-5f);
  float* ob = out + (size_t)wave * CC;
#pragma unroll
  for (int i = 0; i < CC / 32; ++i) { int c = lane + 32 * i; ob[c] = (vals[i] - mu) * inv * g[c] + b[c]; }
}

// ---------------- in-place L2 norm over contiguous rows [rows, 256] ----------------
__global__ void l2n_rows_kernel(float* __restrict__ d, int rows) {
  int wave = (blockIdx.x * blockDim.x + threadIdx.x) >> 5;
  int lane = threadIdx.x & 31;
  if (wave >= rows) return;
  float* r = d + (size_t)wave * CC;
  float s = 0.f;
  float vals[CC / 32];
#pragma unroll
  for (int i = 0; i < CC / 32; ++i) { int c = lane + 32 * i; vals[i] = r[c]; s += vals[i] * vals[i]; }
  float inv = 1.f / fmaxf(sqrtf(wave_sum(s)), 1e-12f);
#pragma unroll
  for (int i = 0; i < CC / 32; ++i) { int c = lane + 32 * i; r[c] = vals[i] * inv; }
}

// ---------------- build [null; tokens] normalized, padded to 32 rows ----------------
__global__ void prep_protos_kernel(const float* __restrict__ nullv, const float* __restrict__ toks,
                                   float* __restrict__ out) {
  int wave = (blockIdx.x * blockDim.x + threadIdx.x) >> 5;
  int lane = threadIdx.x & 31;
  if (wave >= BB * 32) return;
  int bi = wave >> 5, r = wave & 31;
  float* o = out + (size_t)wave * CC;
  int rc = (r >= 1 && r <= KCLS) ? (r - 1) : 0;
  const float* tsrc = toks + ((size_t)bi * KCLS + rc) * CC;
  float vals[CC / 32];
  float s = 0.f;
#pragma unroll
  for (int i = 0; i < CC / 32; ++i) {
    int c = lane + 32 * i;
    float xn = nullv[c];                     // always-safe loads, then select
    float xt = tsrc[c];
    float x = (r == 0) ? xn : ((r <= KCLS) ? xt : 0.f);
    vals[i] = x; s += x * x;
  }
  float inv = 1.f / fmaxf(sqrtf(wave_sum(s)), 1e-12f);
#pragma unroll
  for (int i = 0; i < CC / 32; ++i) { int c = lane + 32 * i; o[c] = vals[i] * inv; }
}

// ---------------- zero-pad prototypes to 32 rows: PROTOP[b,32,PDIM] ----------------
__global__ void pad_proto_kernel(const float* __restrict__ p, float* __restrict__ o) {
  int i = blockIdx.x * blockDim.x + threadIdx.x;
  if (i >= BB * 32 * PDIM) return;
  int d = i & (PDIM - 1);
  int r = (i >> 8) & 31;
  int b = i >> 13;
  int rc = (r < KCLS) ? r : 0;
  float v = p[((size_t)b * KCLS + rc) * PDIM + d];
  o[i] = (r < KCLS) ? v : 0.f;
}

// ---------------- masked softmax over P=32 per (b,h,n) ----------------
__global__ void attn_softmax_kernel(float* __restrict__ sc, const unsigned char* __restrict__ mask) {
  int i = blockIdx.x * blockDim.x + threadIdx.x;
  if (i >= BB * NHEAD * NN) return;
  int bi = i / (NHEAD * NN);
  float* s = sc + (size_t)i * PP;
  const unsigned char* mk = mask + (size_t)bi * PP;
  float v[PP];
  float mx = -3.0e38f;
#pragma unroll
  for (int p = 0; p < PP; ++p) { v[p] = mk[p] ? s[p] : -1.0e9f; mx = fmaxf(mx, v[p]); }
  float den = 0.f;
#pragma unroll
  for (int p = 0; p < PP; ++p) { v[p] = __expf(v[p] - mx); den += v[p]; }
  float inv = 1.f / den;
#pragma unroll
  for (int p = 0; p < PP; ++p) s[p] = v[p] * inv;
}

// ------- class logits -> cw (shifted: l[j] = cw[j+1], j=0..15; l[16..31]=0), presence, pvn -------
__global__ void cls_kernel(float* __restrict__ lg, const unsigned char* __restrict__ cmask,
                           float* __restrict__ pres, float* __restrict__ pvn) {
  int i = blockIdx.x * blockDim.x + threadIdx.x;
  if (i >= BB * NN) return;
  int bi = i / NN;
  float* l = lg + (size_t)i * 32;
  const unsigned char* mk = cmask + (size_t)bi * KCLS;
  float v[KCLS + 1];
  float best = -1.0e4f;
#pragma unroll
  for (int k = 0; k <= KCLS; ++k) v[k] = l[k];
#pragma unroll
  for (int k = 0; k < KCLS; ++k) { float c = mk[k] ? v[k + 1] : -1.0e4f; best = fmaxf(best, c); }
  pvn[i] = best - v[0];
  float mx = v[0];
#pragma unroll
  for (int k = 1; k <= KCLS; ++k) mx = fmaxf(mx, v[k]);
  float den = 0.f;
#pragma unroll
  for (int k = 0; k <= KCLS; ++k) { v[k] = __expf(v[k] - mx); den += v[k]; }
  float inv = 1.f / den;
#pragma unroll
  for (int k = 0; k < KCLS; ++k) l[k] = v[k + 1] * inv;   // shifted class weights
#pragma unroll
  for (int k = KCLS; k < 32; ++k) l[k] = 0.f;
  pres[i] = 1.f - v[0] * inv;
}

// ---------------- scale logits: max_sim, margin, 0.5*(pvn+svn) ----------------
__global__ void slog_kernel(const float* __restrict__ sl, const unsigned char* __restrict__ cmask,
                            const float* __restrict__ pvn, float* __restrict__ maxs,
                            float* __restrict__ marg, float* __restrict__ extra) {
  int i = blockIdx.x * blockDim.x + threadIdx.x;
  if (i >= BB * NN) return;
  int bi = i / NN;
  const float* s = sl + (size_t)i * 32;
  const unsigned char* mk = cmask + (size_t)bi * KCLS;
  float nulls = s[0];
  float t1 = -3.0e38f, t2 = -3.0e38f;
  int cnt = 0;
#pragma unroll
  for (int k = 0; k < KCLS; ++k) {
    float c = mk[k] ? s[k + 1] : -1.0e4f;
    cnt += mk[k] ? 1 : 0;
    if (c > t1) { t2 = t1; t1 = c; } else if (c > t2) { t2 = c; }
  }
  maxs[i] = t1;
  marg[i] = (cnt > 1) ? (t1 - t2) : 1.f;
  extra[i] = 0.5f * (pvn[i] + (t1 - nulls));
}

// ---------------- FiLM: q = q*(1+gamma)+beta ----------------
__global__ void film_kernel(float* __restrict__ q, const float* __restrict__ gb) {
  size_t i = (size_t)blockIdx.x * blockDim.x + threadIdx.x;
  if (i >= (size_t)BB * NN * CC) return;
  size_t row = i / CC; int c = (int)(i % CC);
  float gamma = gb[row * (2 * CC) + c];
  float beta  = gb[row * (2 * CC) + CC + c];
  q[i] = q[i] * (1.f + gamma) + beta;
}

// ---------------- assemble packed-bf16 cat tensor [B*N, 144 pairs] ----------------
__global__ void cat_pack_kernel(const float* __restrict__ q, const float* __restrict__ maxs,
                                const float* __restrict__ marg, const float* __restrict__ pres,
                                const float* __restrict__ extra, unsigned* __restrict__ catp) {
  size_t i = (size_t)blockIdx.x * blockDim.x + threadIdx.x;
  if (i >= (size_t)BB * NN * CP2) return;
  size_t row = i / CP2; int cp = (int)(i % CP2);
  float f0, f1;
  if (cp < CC / 2) {
    const float* qr = q + row * CC + 2 * cp;
    f0 = qr[0]; f1 = qr[1];
  } else if (cp == CC / 2) {
    f0 = maxs[row]; f1 = marg[row];
  } else if (cp == CC / 2 + 1) {
    f0 = pres[row]; f1 = extra[row];
  } else {
    f0 = 0.f; f1 = 0.f;
  }
  catp[i] = pack_bf16_pair(f0, f1);
}

// ------- repack ref_w into bf16-pair layout wp[tap][kpair][o], zero-padded 260->288 -------
__global__ void pack_refw_kernel(const float* __restrict__ w, unsigned* __restrict__ wp) {
  int i = blockIdx.x * blockDim.x + threadIdx.x;
  if (i >= 9 * CP2 * CC) return;
  int o  = i & (CC - 1);
  int kp = (i >> 8) % CP2;
  int t  = i / (CC * CP2);
  int dy = t / 3, dx = t % 3;
  int k0 = 2 * kp, k1 = k0 + 1;
  int k0c = (k0 < CIN) ? k0 : (CIN - 1);
  int k1c = (k1 < CIN) ? k1 : (CIN - 1);
  float f0 = w[(((size_t)o * CIN + k0c) * 3 + dy) * 3 + dx];
  float f1 = w[(((size_t)o * CIN + k1c) * 3 + dy) * 3 + dx];
  f0 = (k0 < CIN) ? f0 : 0.f;
  f1 = (k1 < CIN) ? f1 : 0.f;
  wp[i] = pack_bf16_pair(f0, f1);
}

// ---------------- 3x3 SAME conv as direct implicit WMMA GEMM (all-bf16 operands) ----------------
__global__ void conv3x3_wmma_kernel(const unsigned* __restrict__ catp, const unsigned* __restrict__ wp,
                                    const float* __restrict__ rb, float* __restrict__ out) {
  int wave = (blockIdx.x * blockDim.x + threadIdx.x) >> 5;
  int lane = threadIdx.x & 31;
  const int MT = NN / 16, NT2 = CC / 32;
  if (wave >= BB * MT * NT2) return;
  int n2 = wave % NT2; int tmp = wave / NT2;
  int mi = tmp % MT; int bi = tmp / MT;
  int m = lane & 15, kg = lane >> 4;
  int tok = mi * 16 + m;
  int ty = tok >> 6, tx = tok & 63;
  int o0 = n2 * 32 + (lane & 15);
  const unsigned* catb = catp + (size_t)bi * NN * CP2;

  v8f acc0 = {}; v8f acc1 = {};
  for (int t = 0; t < 9; ++t) {
    int dy = t / 3, dx = t % 3;
    int yy = ty + dy - 1, xx = tx + dx - 1;
    bool valid = ((unsigned)yy < (unsigned)HH) && ((unsigned)xx < (unsigned)WW);
    int cy = min(max(yy, 0), HH - 1), cx = min(max(xx, 0), WW - 1);
    const unsigned* arow = catb + (size_t)(cy * WW + cx) * CP2;
    const unsigned* wrow = wp + (size_t)t * CP2 * CC;
    for (int kc2 = 0; kc2 < CP2; kc2 += 16) {
      v8u au, b0u, b1u;
#pragma unroll
      for (int v = 0; v < 8; ++v) {
        int kp = kc2 + v + 4 * kg + ((v >= 4) ? 4 : 0);
        unsigned d = arow[kp];                    // clamped address; mask value
        au[v] = valid ? d : 0u;
        const unsigned* wk = wrow + (size_t)kp * CC + o0;
        b0u[v] = wk[0];
        b1u[v] = wk[16];
      }
      v16bf a  = __builtin_bit_cast(v16bf, au);
      v16bf b0 = __builtin_bit_cast(v16bf, b0u);
      v16bf b1 = __builtin_bit_cast(v16bf, b1u);
      acc0 = __builtin_amdgcn_wmma_f32_16x16x32_bf16(false, a, false, b0, (short)0, acc0, false, false);
      acc1 = __builtin_amdgcn_wmma_f32_16x16x32_bf16(false, a, false, b1, (short)0, acc1, false, false);
    }
  }
  float* ob = out + (size_t)bi * NN * CC + (size_t)(mi * 16) * CC + n2 * 32;
  int n = lane & 15, mh = lane >> 4;
#pragma unroll
  for (int r = 0; r < 8; ++r) {
    int mm = r + 8 * mh;
    ob[(size_t)mm * CC + n]      = acc0[r] + rb[n2 * 32 + n];
    ob[(size_t)mm * CC + n + 16] = acc1[r] + rb[n2 * 32 + 16 + n];
  }
}

// ---------------- BN + ReLU + transpose to NCHW ----------------
__global__ void bn_relu_out_kernel(const float* __restrict__ conv, const float* __restrict__ g,
                                   const float* __restrict__ b, const float* __restrict__ m,
                                   const float* __restrict__ v, float* __restrict__ out) {
  size_t i = (size_t)blockIdx.x * blockDim.x + threadIdx.x;
  if (i >= (size_t)BB * CC * NN) return;
  int n = (int)(i % NN);
  int c = (int)((i / NN) % CC);
  int bi = (int)(i / ((size_t)NN * CC));
  float val = conv[((size_t)bi * NN + n) * CC + c];
  val = (val - m[c]) * rsqrtf(v[c] + 1e-5f) * g[c] + b[c];
  out[i] = fmaxf(val, 0.f);
}

// ---------------- host side ----------------
static void launch_gemm(const Gemm& g, int bT, int epi, hipStream_t s) {
  long waves = (long)g.outer * g.inner * (g.M / 16) * (g.N / 32);
  unsigned blocks = (unsigned)((waves + 3) / 4);
  if (bT)            gemm_wmma_kernel<1, 0><<<blocks, 128, 0, s>>>(g);
  else if (epi == 0) gemm_wmma_kernel<0, 0><<<blocks, 128, 0, s>>>(g);
  else if (epi == 1) gemm_wmma_kernel<0, 1><<<blocks, 128, 0, s>>>(g);
  else               gemm_wmma_kernel<0, 2><<<blocks, 128, 0, s>>>(g);
}

extern "C" void kernel_launch(void* const* d_in, const int* in_sizes, int n_in,
                              void* d_out, int out_size, void* d_ws, size_t ws_size,
                              hipStream_t stream) {
  (void)in_sizes; (void)n_in; (void)out_size; (void)ws_size;
  const float* x      = (const float*)d_in[0];
  const float* ptk    = (const float*)d_in[1];
  const unsigned char* pmask = (const unsigned char*)d_in[2];
  const float* proto  = (const float*)d_in[3];
  const unsigned char* cmask = (const unsigned char*)d_in[4];
  const float* stok   = (const float*)d_in[5];
  const float* qn_g = (const float*)d_in[6],  *qn_b = (const float*)d_in[7];
  const float* pn_g = (const float*)d_in[8],  *pn_b = (const float*)d_in[9];
  const float* wq = (const float*)d_in[10], *bq = (const float*)d_in[11];
  const float* wk = (const float*)d_in[12], *bk = (const float*)d_in[13];
  const float* wv = (const float*)d_in[14], *bv = (const float*)d_in[15];
  const float* wo = (const float*)d_in[16], *bo = (const float*)d_in[17];
  const float* f1w = (const float*)d_in[18], *f1b = (const float*)d_in[19];
  const float* f2w = (const float*)d_in[20], *f2b = (const float*)d_in[21];
  const float* q2pw = (const float*)d_in[22], *q2pb = (const float*)d_in[23];
  const float* filmw = (const float*)d_in[24], *filmb = (const float*)d_in[25];
  const float* simw = (const float*)d_in[26], *simb = (const float*)d_in[27];
  const float* nullp = (const float*)d_in[28];
  const float* nulls = (const float*)d_in[29];
  const float* refw = (const float*)d_in[30], *refb = (const float*)d_in[31];
  const float* bng = (const float*)d_in[32], *bnb = (const float*)d_in[33];
  const float* bnm = (const float*)d_in[34], *bnv = (const float*)d_in[35];
  float* out = (float*)d_out;
  float* ws = (float*)d_ws;

  // workspace layout (floats)
  const size_t SZ_TOK = (size_t)BB * NN * CC;           // 8.4M
  const size_t O_QLN   = 0;
  const size_t O_QPROJ = O_QLN   + SZ_TOK;
  const size_t O_SCORE = O_QPROJ + SZ_TOK;              // scores / adaptive / conv-out
  const size_t O_HID   = O_SCORE + SZ_TOK;              // 33.5M: hidden / gb / catp
  const size_t O_LOG   = O_HID   + (size_t)BB * NN * 4 * CC;
  const size_t O_PTLN  = O_LOG   + (size_t)BB * NN * 32;
  const size_t O_KB    = O_PTLN  + (size_t)BB * PP * CC;
  const size_t O_VB    = O_KB    + (size_t)BB * PP * CC;
  const size_t O_ALLP  = O_VB    + (size_t)BB * PP * CC;
  const size_t O_ALLS  = O_ALLP  + (size_t)BB * 32 * PDIM;
  const size_t O_PROTOP= O_ALLS  + (size_t)BB * 32 * CC;
  const size_t O_PRES  = O_PROTOP+ (size_t)BB * 32 * PDIM;
  const size_t O_PVN   = O_PRES  + (size_t)BB * NN;
  const size_t O_MAXS  = O_PVN   + (size_t)BB * NN;
  const size_t O_MARG  = O_MAXS  + (size_t)BB * NN;
  const size_t O_EXTRA = O_MARG  + (size_t)BB * NN;
  const size_t O_WPACK = O_EXTRA + (size_t)BB * NN;     // 9*144*256 dwords
  const size_t O_ADAPT = O_SCORE;
  const size_t O_GB    = O_HID;
  const size_t O_CATP  = O_HID;                         // packed bf16 pairs (dwords)
  const size_t O_CONV  = O_SCORE;

  // 1. LayerNorms
  ln_query_kernel<<<(BB * NN) / 8, 256, 0, stream>>>(x, qn_g, qn_b, ws + O_QLN);
  ln_rows_kernel<<<(BB * PP) / 8, 256, 0, stream>>>(ptk, pn_g, pn_b, ws + O_PTLN, BB * PP);

  // 2. Q/K/V projections
  Gemm g{};
  g.outer = 1; g.inner = 1; g.aOuter = g.aInner = g.bOuter = g.bInner = g.cOuter = g.cInner = 0;
  g.scale = 1.f;
  g.A = ws + O_QLN; g.lda = CC; g.Bm = wq; g.ldb = CC; g.C = ws + O_QPROJ; g.ldc = CC;
  g.bias = bq; g.M = BB * NN; g.N = CC; g.K = CC;
  launch_gemm(g, 0, 0, stream);
  g.A = ws + O_PTLN; g.Bm = wk; g.C = ws + O_KB; g.bias = bk; g.M = BB * PP;
  launch_gemm(g, 0, 0, stream);
  g.Bm = wv; g.C = ws + O_VB; g.bias = bv;
  launch_gemm(g, 0, 0, stream);

  // 3. scores[b,h,n,p] = q . k / sqrt(HD)
  g = Gemm{};
  g.outer = BB; g.inner = NHEAD;
  g.A = ws + O_QPROJ; g.aOuter = (long)NN * CC; g.aInner = HD; g.lda = CC;
  g.Bm = ws + O_KB;   g.bOuter = (long)PP * CC; g.bInner = HD; g.ldb = CC;
  g.C = ws + O_SCORE; g.cOuter = (long)NHEAD * NN * PP; g.cInner = (long)NN * PP; g.ldc = PP;
  g.bias = nullptr; g.M = NN; g.N = PP; g.K = HD; g.scale = rsqrtf((float)HD);
  launch_gemm(g, 1, 0, stream);

  // 4. masked softmax over P
  attn_softmax_kernel<<<(BB * NHEAD * NN) / 256, 256, 0, stream>>>(ws + O_SCORE, pmask);

  // 5. attn_out[b,n,h*HD+d] = attn @ v   (overwrites QPROJ)
  g = Gemm{};
  g.outer = BB; g.inner = NHEAD; g.scale = 1.f;
  g.A = ws + O_SCORE; g.aOuter = (long)NHEAD * NN * PP; g.aInner = (long)NN * PP; g.lda = PP;
  g.Bm = ws + O_VB;   g.bOuter = (long)PP * CC; g.bInner = HD; g.ldb = CC;
  g.C = ws + O_QPROJ; g.cOuter = (long)NN * CC; g.cInner = HD; g.ldc = CC;
  g.bias = nullptr; g.M = NN; g.N = HD; g.K = PP;
  launch_gemm(g, 0, 0, stream);

  // 6. residual: query += attn_out @ wo + bo
  g = Gemm{};
  g.outer = 1; g.inner = 1; g.scale = 1.f;
  g.A = ws + O_QPROJ; g.lda = CC; g.Bm = wo; g.ldb = CC;
  g.C = ws + O_QLN; g.ldc = CC; g.bias = bo; g.M = BB * NN; g.N = CC; g.K = CC;
  launch_gemm(g, 0, 2, stream);

  // 7. FFN
  g.A = ws + O_QLN; g.Bm = f1w; g.ldb = 4 * CC; g.C = ws + O_HID; g.ldc = 4 * CC;
  g.bias = f1b; g.N = 4 * CC; g.K = CC;
  launch_gemm(g, 0, 1, stream);
  g.A = ws + O_HID; g.lda = 4 * CC; g.Bm = f2w; g.ldb = CC; g.C = ws + O_QLN; g.ldc = CC;
  g.bias = f2b; g.N = CC; g.K = 4 * CC;
  launch_gemm(g, 0, 2, stream);

  // 8. qp = l2n(query @ q2p_w + b)  (into QPROJ)
  g.A = ws + O_QLN; g.lda = CC; g.Bm = q2pw; g.ldb = PDIM; g.C = ws + O_QPROJ; g.ldc = PDIM;
  g.bias = q2pb; g.N = PDIM; g.K = CC;
  launch_gemm(g, 0, 0, stream);
  l2n_rows_kernel<<<(BB * NN) / 8, 256, 0, stream>>>(ws + O_QPROJ, BB * NN);

  // 9. allp / padded protos
  prep_protos_kernel<<<(BB * 32) / 8, 256, 0, stream>>>(nullp, proto, ws + O_ALLP);
  pad_proto_kernel<<<(BB * 32 * PDIM) / 256, 256, 0, stream>>>(proto, ws + O_PROTOP);

  // 10. logits = qp @ allp^T  (N padded to 32)
  g = Gemm{};
  g.outer = BB; g.inner = 1; g.scale = 1.f;
  g.A = ws + O_QPROJ; g.aOuter = (long)NN * PDIM; g.lda = PDIM;
  g.Bm = ws + O_ALLP; g.bOuter = (long)32 * PDIM; g.ldb = PDIM;
  g.C = ws + O_LOG;   g.cOuter = (long)NN * 32;   g.ldc = 32;
  g.bias = nullptr; g.M = NN; g.N = 32; g.K = PDIM;
  launch_gemm(g, 1, 0, stream);

  // 11. softmax(17) -> shifted cw (in place), presence, pvn
  cls_kernel<<<(BB * NN) / 256, 256, 0, stream>>>(ws + O_LOG, cmask, ws + O_PRES, ws + O_PVN);

  // 12. adaptive = cw_cls @ proto_padded  (K padded to 32, into SCORE buffer)
  g = Gemm{};
  g.outer = BB; g.inner = 1; g.scale = 1.f;
  g.A = ws + O_LOG; g.aOuter = (long)NN * 32; g.lda = 32;
  g.Bm = ws + O_PROTOP; g.bOuter = (long)32 * PDIM; g.ldb = PDIM;
  g.C = ws + O_ADAPT; g.cOuter = (long)NN * PDIM; g.ldc = PDIM;
  g.bias = nullptr; g.M = NN; g.N = PDIM; g.K = 32;
  launch_gemm(g, 0, 0, stream);

  // 13. gb = adaptive @ film_w + film_b  (into HID region)
  g = Gemm{};
  g.outer = 1; g.inner = 1; g.scale = 1.f;
  g.A = ws + O_ADAPT; g.lda = PDIM; g.Bm = filmw; g.ldb = 2 * CC;
  g.C = ws + O_GB; g.ldc = 2 * CC; g.bias = filmb;
  g.M = BB * NN; g.N = 2 * CC; g.K = PDIM;
  launch_gemm(g, 0, 0, stream);

  // 14. FiLM in place on query
  film_kernel<<<(unsigned)((SZ_TOK + 255) / 256), 256, 0, stream>>>(ws + O_QLN, ws + O_GB);

  // 15. sim_feat = query @ sim_w^T + sim_b  (into QPROJ), then l2n
  g.A = ws + O_QLN; g.lda = CC; g.Bm = simw; g.ldb = CC;
  g.C = ws + O_QPROJ; g.ldc = CC; g.bias = simb; g.N = CC; g.K = CC;
  launch_gemm(g, 1, 0, stream);
  l2n_rows_kernel<<<(BB * NN) / 8, 256, 0, stream>>>(ws + O_QPROJ, BB * NN);

  // 16. alls = [l2n(null_scale); l2n(scale_tokens)]
  prep_protos_kernel<<<(BB * 32) / 8, 256, 0, stream>>>(nulls, stok, ws + O_ALLS);

  // 17. slog = feat @ alls^T  (into LOG buffer)
  g = Gemm{};
  g.outer = BB; g.inner = 1; g.scale = 1.f;
  g.A = ws + O_QPROJ; g.aOuter = (long)NN * CC; g.lda = CC;
  g.Bm = ws + O_ALLS; g.bOuter = (long)32 * CC; g.ldb = CC;
  g.C = ws + O_LOG; g.cOuter = (long)NN * 32; g.ldc = 32;
  g.bias = nullptr; g.M = NN; g.N = 32; g.K = CC;
  launch_gemm(g, 1, 0, stream);

  // 18. max_sim / margin / 0.5*(pvn+svn)
  slog_kernel<<<(BB * NN) / 256, 256, 0, stream>>>(ws + O_LOG, cmask, ws + O_PVN,
                                                   ws + O_MAXS, ws + O_MARG, ws + O_EXTRA);

  // 19. packed cat tensor (bf16 pairs) + packed conv weights
  cat_pack_kernel<<<(unsigned)(((size_t)BB * NN * CP2 + 255) / 256), 256, 0, stream>>>(
      ws + O_QLN, ws + O_MAXS, ws + O_MARG, ws + O_PRES, ws + O_EXTRA,
      (unsigned*)(ws + O_CATP));
  pack_refw_kernel<<<(9 * CP2 * CC) / 256, 256, 0, stream>>>(refw, (unsigned*)(ws + O_WPACK));

  // 20. 3x3 conv (implicit WMMA GEMM) + ref_b  (into SCORE buffer)
  {
    unsigned waves = BB * (NN / 16) * (CC / 32);
    conv3x3_wmma_kernel<<<(waves + 3) / 4, 128, 0, stream>>>(
        (const unsigned*)(ws + O_CATP), (const unsigned*)(ws + O_WPACK), refb, ws + O_CONV);
  }

  // 21. BN + ReLU + NCHW transpose to output
  bn_relu_out_kernel<<<(unsigned)((SZ_TOK + 255) / 256), 256, 0, stream>>>(
      ws + O_CONV, bng, bnb, bnm, bnv, out);
}
